// MultiHeadAttention_5643587027068
// MI455X (gfx1250) — matrix operand order repaired and checked by reading, not compiled, not run on hardware
//
#include <hip/hip_runtime.h>

#define BB 4
#define SS 2048
#define DD 1024
#define HH 16
#define DH 64

typedef __attribute__((ext_vector_type(16))) __bf16        bf16x16;
typedef __attribute__((ext_vector_type(8)))  float         f32x8;
typedef __attribute__((ext_vector_type(4)))  float         f32x4;
typedef __attribute__((ext_vector_type(4)))  unsigned int  u32x4;

union AFrag {
    bf16x16 v;
    unsigned int u[8];
    u32x4 q[2];
};

__device__ __forceinline__ unsigned short f2bf(float f) {
    unsigned int u = __builtin_bit_cast(unsigned int, f);
    u += 0x7FFFu + ((u >> 16) & 1u);   // round-to-nearest-even
    return (unsigned short)(u >> 16);
}
__device__ __forceinline__ unsigned int pack2(float lo, float hi) {
    return (unsigned int)f2bf(lo) | ((unsigned int)f2bf(hi) << 16);
}

// ---------------------------------------------------------------------------
// Kernel 1: W (f32, [D,D]) -> bf16 workspace copy
// ---------------------------------------------------------------------------
__global__ __launch_bounds__(256) void convw_kernel(const float* __restrict__ W,
                                                    unsigned int* __restrict__ Wb) {
    unsigned idx = blockIdx.x * 256 + threadIdx.x;      // vec4 index, DD*DD/4 total
    f32x4 v = *(const f32x4*)(W + (size_t)idx * 4);
    Wb[idx * 2]     = pack2(v.x, v.y);
    Wb[idx * 2 + 1] = pack2(v.z, v.w);
}

// ---------------------------------------------------------------------------
// Kernel 2: flash attention per (b,h).  Block = 256 thr = 8 waves,
// each wave owns 16 query rows (block tile = 128 q rows).
// K/V tiles double-buffered in LDS; next tile's global loads are issued
// before the current tile's WMMAs so VMEM latency hides behind compute.
// ---------------------------------------------------------------------------
__global__ __launch_bounds__(256) void attn_kernel(const float* __restrict__ Q,
                                                   const float* __restrict__ K,
                                                   const float* __restrict__ V,
                                                   const float* __restrict__ mask,
                                                   unsigned short* __restrict__ Xb) {
    __shared__ __align__(16) unsigned int   sQ[128 * 32];        // 128 q x 64 d (bf16 pairs)
    __shared__ __align__(16) unsigned int   sK[2][64 * 32];      // 64 keys x 64 d, 2 buffers
    __shared__ __align__(16) unsigned short sVT[2][64 * 72];     // [d][key] pad 72, 2 buffers
    __shared__ __align__(16) unsigned short sP[8 * 16 * 32];     // per-wave P staging

    const int tid  = threadIdx.x;
    const int wave = tid >> 5;
    const int lane = tid & 31;
    const int lg   = lane & 15;   // lane within 16-group
    const int hi   = lane >> 4;   // which half of the wave

    const int bh = blockIdx.z * HH + blockIdx.y;
    const size_t base = (size_t)bh * SS * DH;   // contiguous per-head block (raw reshape)
    const float* Qg = Q + base;

    const int qblk = blockIdx.x * 128;
    const int q0   = qblk + wave * 16;
    const float scale = 0.125f;                 // 1/sqrt(64)
    const float L2E   = 1.44269504088896f;

    // per-thread staging coordinates (constant across tiles)
    const int trow = tid >> 4;                  // 0..15
    const int tc4  = tid & 15;
    const float* kptr = K + base + (size_t)trow * DH + tc4 * 4;
    const float* vptr = V + base + (size_t)trow * DH + tc4 * 4;

    // ---- stage Q block (128x64) as bf16 in LDS ----
    for (int it = 0; it < 8; ++it) {
        int row = trow + it * 16;               // 128 rows total
        f32x4 v = *(const f32x4*)(Qg + (size_t)(qblk + row) * DH + tc4 * 4);
        sQ[row * 32 + tc4 * 2]     = pack2(v.x, v.y);
        sQ[row * 32 + tc4 * 2 + 1] = pack2(v.z, v.w);
    }

    // ---- prologue: stage K/V tile 0 into buffer 0 ----
    {
        f32x4 gk[4], gv[4];
        for (int it = 0; it < 4; ++it) {
            gk[it] = *(const f32x4*)(kptr + it * 16 * DH);
            gv[it] = *(const f32x4*)(vptr + it * 16 * DH);
        }
        for (int it = 0; it < 4; ++it) {
            int row = trow + it * 16;
            sK[0][row * 32 + tc4 * 2]     = pack2(gk[it].x, gk[it].y);
            sK[0][row * 32 + tc4 * 2 + 1] = pack2(gk[it].z, gk[it].w);
            int d = tc4 * 4;
            sVT[0][(d + 0) * 72 + row] = f2bf(gv[it].x);
            sVT[0][(d + 1) * 72 + row] = f2bf(gv[it].y);
            sVT[0][(d + 2) * 72 + row] = f2bf(gv[it].z);
            sVT[0][(d + 3) * 72 + row] = f2bf(gv[it].w);
        }
    }
    __syncthreads();

    // ---- Q A-fragments (16x32 each, d-blocks 0-31 / 32-63), live whole loop ----
    AFrag aq[2];
    {
        const u32x4* sQ4 = (const u32x4*)sQ;    // row = 8 units
        int row = wave * 16 + lg;
        for (int kb = 0; kb < 2; ++kb) {
            aq[kb].q[0] = sQ4[row * 8 + kb * 4 + hi * 2];
            aq[kb].q[1] = sQ4[row * 8 + kb * 4 + hi * 2 + 1];
        }
    }

    // online-softmax state; vgpr slot i holds row (q0 + i + 8*hi)
    float mrow[8], lrow[8];
    f32x8 acc[4] = {};
    unsigned mrowoff[8];                        // 32-bit mask row offsets (saddr mode)
    for (int i = 0; i < 8; ++i) {
        mrow[i] = -3.0e38f;
        lrow[i] = 0.0f;
        mrowoff[i] = (unsigned)(q0 + i + 8 * hi) * (unsigned)SS;
    }

    unsigned short* sPw = sP + wave * (16 * 32);
    const u32x4* sP4 = (const u32x4*)sPw;

    int cur = 0;
#pragma unroll 1
    for (int kt = 0; kt < SS; kt += 64) {
        const bool have_next = (kt + 64 < SS);
        // ---- issue next tile's global loads (latency overlapped w/ compute) ----
        f32x4 gk[4], gv[4];
        if (have_next) {
            const float* kp = kptr + (size_t)(kt + 64) * DH;
            const float* vp = vptr + (size_t)(kt + 64) * DH;
            for (int it = 0; it < 4; ++it) {
                gk[it] = *(const f32x4*)(kp + it * 16 * DH);
                gv[it] = *(const f32x4*)(vp + it * 16 * DH);
            }
        }

        const u32x4* sK4 = (const u32x4*)(&sK[cur][0]);
        const unsigned short* sVTc = &sVT[cur][0];

        // ---- two 32-key chunks per 64-key tile ----
#pragma unroll
        for (int h32 = 0; h32 < 2; ++h32) {
            const int kbase = h32 * 32;

            // load all four B fragments up front, then 4 back-to-back WMMAs
            AFrag bk00, bk01, bk10, bk11;       // [t][kb]
            {
                int kr0 = (kbase + lg) * 8;
                int kr1 = (kbase + 16 + lg) * 8;
                bk00.q[0] = sK4[kr0 + hi * 2];     bk00.q[1] = sK4[kr0 + hi * 2 + 1];
                bk01.q[0] = sK4[kr0 + 4 + hi * 2]; bk01.q[1] = sK4[kr0 + 4 + hi * 2 + 1];
                bk10.q[0] = sK4[kr1 + hi * 2];     bk10.q[1] = sK4[kr1 + hi * 2 + 1];
                bk11.q[0] = sK4[kr1 + 4 + hi * 2]; bk11.q[1] = sK4[kr1 + 4 + hi * 2 + 1];
            }
            f32x8 c0 = {}, c1 = {};
            c0 = __builtin_amdgcn_wmma_f32_16x16x32_bf16(false, aq[0].v, false, bk00.v,
                                                         (short)0, c0, false, false);
            c1 = __builtin_amdgcn_wmma_f32_16x16x32_bf16(false, aq[0].v, false, bk10.v,
                                                         (short)0, c1, false, false);
            c0 = __builtin_amdgcn_wmma_f32_16x16x32_bf16(false, aq[1].v, false, bk01.v,
                                                         (short)0, c0, false, false);
            c1 = __builtin_amdgcn_wmma_f32_16x16x32_bf16(false, aq[1].v, false, bk11.v,
                                                         (short)0, c1, false, false);

            // scale + mask (32-bit offsets -> saddr addressing)
            const unsigned kc0 = (unsigned)(kt + kbase + lg);
            const unsigned kc1 = kc0 + 16u;
            float corr[8];
#pragma unroll
            for (int i = 0; i < 8; ++i) {
                float s0 = c0[i] * scale + mask[mrowoff[i] + kc0];
                float s1 = c1[i] * scale + mask[mrowoff[i] + kc1];
                float t = fmaxf(s0, s1);
                t = fmaxf(t, __shfl_xor(t, 1, 32));
                t = fmaxf(t, __shfl_xor(t, 2, 32));
                t = fmaxf(t, __shfl_xor(t, 4, 32));
                t = fmaxf(t, __shfl_xor(t, 8, 32));
                float mnew = fmaxf(mrow[i], t);
                float cr = __builtin_amdgcn_exp2f((mrow[i] - mnew) * L2E);
                float p0 = __builtin_amdgcn_exp2f((s0 - mnew) * L2E);
                float p1 = __builtin_amdgcn_exp2f((s1 - mnew) * L2E);
                c0[i] = p0;
                c1[i] = p1;
                float rs = p0 + p1;
                rs += __shfl_xor(rs, 1, 32);
                rs += __shfl_xor(rs, 2, 32);
                rs += __shfl_xor(rs, 4, 32);
                rs += __shfl_xor(rs, 8, 32);
                lrow[i] = lrow[i] * cr + rs;
                mrow[i] = mnew;
                corr[i] = cr;
            }
#pragma unroll
            for (int dt = 0; dt < 4; ++dt)
#pragma unroll
                for (int i = 0; i < 8; ++i) acc[dt][i] *= corr[i];

            // C-layout -> A-layout for P via per-wave LDS (same-wave LDS is in-order)
#pragma unroll
            for (int i = 0; i < 8; ++i) {
                sPw[(i + 8 * hi) * 32 + lg]      = f2bf(c0[i]);
                sPw[(i + 8 * hi) * 32 + 16 + lg] = f2bf(c1[i]);
            }
            AFrag ap;
            ap.q[0] = sP4[lg * 4 + hi * 2];
            ap.q[1] = sP4[lg * 4 + hi * 2 + 1];

            // PV: load all four B fragments, then 4 in-place WMMAs
            AFrag bv[4];
#pragma unroll
            for (int dt = 0; dt < 4; ++dt) {
                const unsigned short* src = &sVTc[(dt * 16 + lg) * 72 + kbase + hi * 16];
                bv[dt].q[0] = *(const u32x4*)(src);
                bv[dt].q[1] = *(const u32x4*)(src + 8);
            }
#pragma unroll
            for (int dt = 0; dt < 4; ++dt)
                acc[dt] = __builtin_amdgcn_wmma_f32_16x16x32_bf16(false, ap.v, false, bv[dt].v,
                                                                  (short)0, acc[dt], false, false);
        }

        // ---- convert & store next tile into the other buffer, then barrier ----
        if (have_next) {
            int nxt = cur ^ 1;
            for (int it = 0; it < 4; ++it) {
                int row = trow + it * 16;
                sK[nxt][row * 32 + tc4 * 2]     = pack2(gk[it].x, gk[it].y);
                sK[nxt][row * 32 + tc4 * 2 + 1] = pack2(gk[it].z, gk[it].w);
                int d = tc4 * 4;
                sVT[nxt][(d + 0) * 72 + row] = f2bf(gv[it].x);
                sVT[nxt][(d + 1) * 72 + row] = f2bf(gv[it].y);
                sVT[nxt][(d + 2) * 72 + row] = f2bf(gv[it].z);
                sVT[nxt][(d + 3) * 72 + row] = f2bf(gv[it].w);
            }
        }
        __syncthreads();
        cur ^= 1;
    }

    // ---- epilogue: x = acc / l as bf16, raw-reshape flat layout ----
    float inv[8];
#pragma unroll
    for (int i = 0; i < 8; ++i) inv[i] = __builtin_amdgcn_rcpf(lrow[i]);
    unsigned short* xhead = Xb + base;
#pragma unroll
    for (int dt = 0; dt < 4; ++dt) {
        int d = dt * 16 + lg;
#pragma unroll
        for (int i = 0; i < 8; ++i) {
            int qrow = q0 + i + 8 * hi;
            xhead[(size_t)qrow * DH + d] = f2bf(acc[dt][i] * inv[i]);
        }
    }
}

// ---------------------------------------------------------------------------
// Kernel 3: out[m,n] = sum_k x[m,k] * W[n,k] + b[n]   (M=8192, N=K=1024)
// One wave computes a 16x64 output tile; W rows load directly as B fragments.
// 32-bit element offsets -> SGPR-base + VGPR-offset addressing.
// ---------------------------------------------------------------------------
__global__ __launch_bounds__(256) void gemm_kernel(const unsigned short* __restrict__ Xb,
                                                   const unsigned short* __restrict__ Wb,
                                                   const float* __restrict__ bias,
                                                   float* __restrict__ out) {
    const int tid  = threadIdx.x;
    const int wave = tid >> 5;
    const int lane = tid & 31;
    const int lg   = lane & 15;
    const int hi   = lane >> 4;

    const int fw    = blockIdx.x * 8 + wave;
    const int mtile = fw >> 4;                  // 512 m-tiles
    const int ntile = fw & 15;                  // 16 n-tiles
    const int m0 = mtile * 16, n0 = ntile * 64;

    const u32x4* X4 = (const u32x4*)Xb;         // row = 128 units of 8 bf16
    const u32x4* W4 = (const u32x4*)Wb;

    f32x8 acc[4] = {};
    const unsigned arow = (unsigned)(m0 + lg) * 128u;
    const unsigned brow0 = (unsigned)(n0 + lg) * 128u;

#pragma unroll 2
    for (int k0 = 0; k0 < DD; k0 += 32) {
        const unsigned ku = (unsigned)(k0 >> 3);
        AFrag a;
        a.q[0] = X4[arow + ku + hi * 2];
        a.q[1] = X4[arow + ku + hi * 2 + 1];
        AFrag b[4];
#pragma unroll
        for (int nt = 0; nt < 4; ++nt) {
            unsigned bidx = brow0 + (unsigned)(nt * 16 * 128) + ku + hi * 2;
            b[nt].q[0] = W4[bidx];
            b[nt].q[1] = W4[bidx + 1];
        }
#pragma unroll
        for (int nt = 0; nt < 4; ++nt)
            acc[nt] = __builtin_amdgcn_wmma_f32_16x16x32_bf16(false, a.v, false, b[nt].v,
                                                              (short)0, acc[nt], false, false);
    }

#pragma unroll
    for (int nt = 0; nt < 4; ++nt) {
        int n = n0 + nt * 16 + lg;
        float bv = bias[n];
#pragma unroll
        for (int i = 0; i < 8; ++i) {
            int m = m0 + i + 8 * hi;
            out[(size_t)m * DD + n] = acc[nt][i] + bv;
        }
    }
}

// ---------------------------------------------------------------------------
extern "C" void kernel_launch(void* const* d_in, const int* in_sizes, int n_in,
                              void* d_out, int out_size, void* d_ws, size_t ws_size,
                              hipStream_t stream) {
    const float* Q    = (const float*)d_in[0];
    const float* K    = (const float*)d_in[1];
    const float* V    = (const float*)d_in[2];
    const float* mask = (const float*)d_in[3];
    const float* W    = (const float*)d_in[4];
    const float* bias = (const float*)d_in[5];
    float* out = (float*)d_out;

    unsigned short* Xb = (unsigned short*)d_ws;                         // 16 MB bf16 x
    unsigned int*   Wb = (unsigned int*)((char*)d_ws + (size_t)BB * SS * DD * 2);

    convw_kernel<<<(DD * DD / 4) / 256, 256, 0, stream>>>(W, Wb);

    dim3 ag(SS / 128, HH, BB);
    attn_kernel<<<ag, 256, 0, stream>>>(Q, K, V, mask, Xb);

    gemm_kernel<<<(DD / 64) * (BB * SS / 16) / 8, 256, 0, stream>>>(
        Xb, (const unsigned short*)Wb, bias, out);
}
